// GPTBlock_46763603919345
// MI455X (gfx1250) — matrix operand order repaired and checked by reading, not compiled, not run on hardware
//
#include <hip/hip_runtime.h>
#include <hip/hip_bf16.h>
#include <math.h>
#include <stdint.h>

// GPT block: pre-LN attention + FFN, fp32 in/out, bf16 WMMA internally.
// B=2, S=2048, D=1024, H=16, d=64, FF=4096.
// GEMMs: async-to-LDS double-buffered pipeline (ASYNCcnt) feeding
// v_wmma_f32_16x16x32_bf16. Attention: per-wave flash attention.

#define EMB 1024
#define HEADS 16
#define HEAD_DIM 64
#define SEQ 2048
#define BATCH 2
#define FFD 4096
#define NROWS (BATCH * SEQ)   // 4096 token rows

#define TILE_M 128
#define TILE_N 128
#define KSTAGE 64             // K depth per LDS stage (2 WMMA k-steps)
#define LDS_LD 72             // padded row length (bf16) -> 144B row stride

typedef __bf16 bf16_t;
typedef __attribute__((ext_vector_type(16))) __bf16 v16bf;
typedef __attribute__((ext_vector_type(8)))  __bf16 v8bf;
typedef __attribute__((ext_vector_type(8)))  float  v8f;

union ABFrag { v16bf v; v8bf part[2]; };

// ---------------------------------------------------------------------------
// WMMA fragment loaders (CDNA5 ISA §7.12.2, wave32)
// A (16x32, MxK): lane 0..15 = row M=l, elems 0..7 -> K 0..7, 8..15 -> K 16..23;
//                 lanes 16..31 take the K 8..15 / 24..31 halves.
// B (32x16, KxN): lane = col N=l&15, elems 0..7 -> K half*8.., 8..15 -> K 16+half*8..
//                 Source must be row-major [N, K].
// C/D (16x16 f32): VGPR j -> M = j + half*8, lane -> N = l&15.
// ---------------------------------------------------------------------------
__device__ __forceinline__ v16bf frag_load_a(const bf16_t* base, int ld) {
  const int lane = threadIdx.x & 31;
  const int row  = lane & 15;
  const int half = lane >> 4;
  const bf16_t* p = base + (size_t)row * ld + half * 8;
  ABFrag f;
  f.part[0] = *(const v8bf*)(p);
  f.part[1] = *(const v8bf*)(p + 16);
  return f.v;
}

__device__ __forceinline__ v16bf frag_load_b(const bf16_t* base, int ld) {
  const int lane = threadIdx.x & 31;
  const int col  = lane & 15;
  const int half = lane >> 4;
  const bf16_t* p = base + (size_t)col * ld + half * 8;
  ABFrag f;
  f.part[0] = *(const v8bf*)(p);
  f.part[1] = *(const v8bf*)(p + 16);
  return f.v;
}

__device__ __forceinline__ v8f wmma_bf16(v16bf a, v16bf b, v8f c) {
  // (neg_a, A, neg_b, B, c_mod, C, reuse_a, reuse_b)
  return __builtin_amdgcn_wmma_f32_16x16x32_bf16(false, a, false, b,
                                                 (short)0, c, false, false);
}

// Async copy of one 16B chunk per lane into LDS (GV addressing mode).
__device__ __forceinline__ void async_b128(uint32_t lds_off, const void* gaddr) {
  asm volatile("global_load_async_to_lds_b128 %0, %1, off"
               :: "v"(lds_off), "v"(gaddr) : "memory");
}

// Stage a 128x64 bf16 tile of A and of B into LDS (8 async issues/thread).
__device__ __forceinline__ void gemm_stage_async(
    const bf16_t* gA, int lda, const bf16_t* gB, int ldb, int k,
    bf16_t (*as)[LDS_LD], bf16_t (*bs)[LDS_LD], int tid) {
#pragma unroll
  for (int i = 0; i < 4; ++i) {
    const int t   = i * 256 + tid;   // 0..1023 lane-chunks
    const int r   = t >> 3;          // tile row 0..127
    const int seg = (t & 7) << 3;    // element offset 0,8,..,56
    async_b128((uint32_t)(uintptr_t)&as[r][seg],
               gA + (size_t)r * lda + k + seg);
    async_b128((uint32_t)(uintptr_t)&bs[r][seg],
               gB + (size_t)r * ldb + k + seg);
  }
}

// ---------------------------------------------------------------------------
// fp32 [K,N] -> bf16 [N,K] transpose (once per launch; weights live in L2)
// ---------------------------------------------------------------------------
__global__ void __launch_bounds__(256)
cvt_transpose_kernel(const float* __restrict__ W, bf16_t* __restrict__ Wt,
                     int K, int N) {
  size_t idx = (size_t)blockIdx.x * 256 + threadIdx.x;
  if (idx >= (size_t)K * N) return;
  int n = (int)(idx / K);
  int k = (int)(idx % K);
  Wt[idx] = (bf16_t)W[(size_t)k * N + n];
}

// ---------------------------------------------------------------------------
// LayerNorm over EMB=1024, one 256-thread block per row, bf16 output
// ---------------------------------------------------------------------------
__global__ void __launch_bounds__(256)
layernorm_kernel(const float* __restrict__ X, const float* __restrict__ g,
                 const float* __restrict__ b, bf16_t* __restrict__ out) {
  const int row = blockIdx.x;
  const float* x = X + (size_t)row * EMB;
  float v[4];
  float s = 0.f, s2 = 0.f;
#pragma unroll
  for (int i = 0; i < 4; ++i) {
    int c = i * 256 + threadIdx.x;
    v[i] = x[c];
    s += v[i];
    s2 += v[i] * v[i];
  }
#pragma unroll
  for (int o = 16; o > 0; o >>= 1) {
    s  += __shfl_xor(s, o, 32);
    s2 += __shfl_xor(s2, o, 32);
  }
  __shared__ float ls[8], ls2[8];
  const int wid = threadIdx.x >> 5;
  if ((threadIdx.x & 31) == 0) { ls[wid] = s; ls2[wid] = s2; }
  __syncthreads();
  float S = 0.f, S2 = 0.f;
#pragma unroll
  for (int w = 0; w < 8; ++w) { S += ls[w]; S2 += ls2[w]; }
  const float mu  = S * (1.f / EMB);
  const float var = S2 * (1.f / EMB) - mu * mu;
  const float inv = rsqrtf(var + 1e-5f);
#pragma unroll
  for (int i = 0; i < 4; ++i) {
    int c = i * 256 + threadIdx.x;
    out[(size_t)row * EMB + c] = (bf16_t)((v[i] - mu) * inv * g[c] + b[c]);
  }
}

// ---------------------------------------------------------------------------
// TN GEMM: C[M,N] = A[M,K] (bf16 row-major) x Bt[N,K]^T (bf16 row-major)
// 256 threads = 8 waves (2Mx4N); wave tile 64x32; block tile 128x128.
// Double-buffered LDS stages of 128x64 per operand filled by
// global_load_async_to_lds_b128; pipelined with s_wait_asynccnt.
// MODE 0: bf16 store          MODE 1: bf16 store, V transposed per head
// MODE 2: fp32 = acc+bias+res MODE 3: bf16 = gelu(acc+bias)
// ---------------------------------------------------------------------------
template <int MODE>
__global__ void __launch_bounds__(256)
gemm_tn_kernel(const bf16_t* __restrict__ A, const bf16_t* __restrict__ Bt,
               int K, int lda, int ldb,
               const float* __restrict__ bias, const float* __restrict__ resid,
               void* __restrict__ outp, int ldc) {
  __shared__ alignas(16) bf16_t aS[2][TILE_M][LDS_LD];  // 2 x 18KB
  __shared__ alignas(16) bf16_t bS[2][TILE_N][LDS_LD];  // 2 x 18KB
  const int tid = threadIdx.x;
  const int wid = tid >> 5;
  const int wm  = wid & 1;
  const int wn  = wid >> 1;
  const int m0b = blockIdx.y * TILE_M;
  const int n0b = blockIdx.x * TILE_N;
  const bf16_t* gA = A + (size_t)m0b * lda;
  const bf16_t* gB = Bt + (size_t)n0b * ldb;

  v8f acc[4][2];
#pragma unroll
  for (int mi = 0; mi < 4; ++mi)
#pragma unroll
    for (int ni = 0; ni < 2; ++ni)
#pragma unroll
      for (int j = 0; j < 8; ++j) acc[mi][ni][j] = 0.f;

  const int KT = K >> 6;  // stages of 64
  gemm_stage_async(gA, lda, gB, ldb, 0, aS[0], bS[0], tid);

  for (int kt = 0; kt < KT; ++kt) {
    const int cur = kt & 1;
    if (kt + 1 < KT) {
      gemm_stage_async(gA, lda, gB, ldb, (kt + 1) << 6,
                       aS[cur ^ 1], bS[cur ^ 1], tid);
      // 8 outstanding from next stage; current stage's 8 have retired
      asm volatile("s_wait_asynccnt 8" ::: "memory");
    } else {
      asm volatile("s_wait_asynccnt 0" ::: "memory");
    }
    __syncthreads();

#pragma unroll
    for (int kk = 0; kk < KSTAGE; kk += 32) {
      v16bf a[4], b[2];
#pragma unroll
      for (int mi = 0; mi < 4; ++mi)
        a[mi] = frag_load_a(&aS[cur][wm * 64 + mi * 16][kk], LDS_LD);
#pragma unroll
      for (int ni = 0; ni < 2; ++ni)
        b[ni] = frag_load_b(&bS[cur][wn * 32 + ni * 16][kk], LDS_LD);
#pragma unroll
      for (int mi = 0; mi < 4; ++mi)
#pragma unroll
        for (int ni = 0; ni < 2; ++ni)
          acc[mi][ni] = wmma_bf16(a[mi], b[ni], acc[mi][ni]);
    }
    __syncthreads();
  }

  const int lane = tid & 31;
  const int nn   = lane & 15;
  const int half = lane >> 4;
  const int m0   = m0b + wm * 64;
  const int n0   = n0b + wn * 32;
#pragma unroll
  for (int mi = 0; mi < 4; ++mi)
#pragma unroll
    for (int ni = 0; ni < 2; ++ni)
#pragma unroll
      for (int j = 0; j < 8; ++j) {
        const int m = m0 + mi * 16 + half * 8 + j;
        const int n = n0 + ni * 16 + nn;
        const float vv = acc[mi][ni][j];
        if constexpr (MODE == 0) {
          ((bf16_t*)outp)[(size_t)m * ldc + n] = (bf16_t)vv;
        } else if constexpr (MODE == 1) {
          // V projection: bf16 transposed per head -> Vt[b][h][d][s]
          const int bb = m / SEQ, ss = m % SEQ;
          const int hh = n >> 6, dd = n & 63;
          ((bf16_t*)outp)[(((size_t)bb * HEADS + hh) * HEAD_DIM + dd) * SEQ + ss] =
              (bf16_t)vv;
        } else if constexpr (MODE == 2) {
          ((float*)outp)[(size_t)m * ldc + n] =
              vv + bias[n] + resid[(size_t)m * ldc + n];
        } else {
          const float xg = vv + bias[n];
          const float ge = 0.5f * xg * (1.f + erff(xg * 0.70710678118654752f));
          ((bf16_t*)outp)[(size_t)m * ldc + n] = (bf16_t)ge;
        }
      }
}

// ---------------------------------------------------------------------------
// Flash attention: one wave per 32(q) x 64(d) output tile of one (b,h).
// Streams causal K-tiles of 64 columns: S=Q*K^T via WMMA, online softmax in
// C-fragment layout, P staged through LDS to re-shape into A fragments,
// O += P*V via WMMA.  Block = 128 threads = 4 independent waves.
// Causal masking only applied to the (single) diagonal-straddling tile.
// Ob may alias Qb (each wave writes exactly the region only it reads).
// ---------------------------------------------------------------------------
__global__ void __launch_bounds__(128)
attention_kernel(const bf16_t* __restrict__ Qb, const bf16_t* __restrict__ Kb,
                 const bf16_t* __restrict__ Vt, bf16_t* Ob) {
  __shared__ alignas(16) bf16_t p_lds[4][32][72];  // per-wave P tile, padded

  const int wid  = threadIdx.x >> 5;
  const int lane = threadIdx.x & 31;
  const int half = lane >> 4;
  const int l15  = lane & 15;

  const int gw = blockIdx.x * 4 + wid;       // global wave index
  const int qt = gw & (SEQ / 32 - 1);        // q tile (64 per b,h)
  const int bh = gw >> 6;
  const int h  = bh & (HEADS - 1);
  const int b  = bh >> 4;
  const int q0 = qt * 32;

  const bf16_t* Qbase = Qb + ((size_t)b * SEQ) * EMB + h * HEAD_DIM;
  const bf16_t* Kbase = Kb + ((size_t)b * SEQ) * EMB + h * HEAD_DIM;
  const bf16_t* Vbase = Vt + ((size_t)(b * HEADS + h) * HEAD_DIM) * SEQ;

  v8f acc[2][4];
  float mrow[2][8], lrow[2][8];
#pragma unroll
  for (int mi = 0; mi < 2; ++mi)
#pragma unroll
    for (int j = 0; j < 8; ++j) {
      mrow[mi][j] = -3.0e38f;
      lrow[mi][j] = 0.f;
    }
#pragma unroll
  for (int mi = 0; mi < 2; ++mi)
#pragma unroll
    for (int dn = 0; dn < 4; ++dn)
#pragma unroll
      for (int j = 0; j < 8; ++j) acc[mi][dn][j] = 0.f;

  for (int kb = 0; kb < q0 + 32; kb += 64) {
    // ---- S = Q x K^T (32x64 tile, inner dim d=64 -> two WMMA k-steps) ----
    v8f sf[2][4];
#pragma unroll
    for (int mi = 0; mi < 2; ++mi)
#pragma unroll
      for (int ni = 0; ni < 4; ++ni)
#pragma unroll
        for (int j = 0; j < 8; ++j) sf[mi][ni][j] = 0.f;

#pragma unroll
    for (int kk = 0; kk < 64; kk += 32) {
      v16bf qf[2], kf[4];
#pragma unroll
      for (int mi = 0; mi < 2; ++mi)
        qf[mi] = frag_load_a(Qbase + (size_t)(q0 + mi * 16) * EMB + kk, EMB);
#pragma unroll
      for (int ni = 0; ni < 4; ++ni)
        kf[ni] = frag_load_b(Kbase + (size_t)(kb + ni * 16) * EMB + kk, EMB);
#pragma unroll
      for (int mi = 0; mi < 2; ++mi)
#pragma unroll
        for (int ni = 0; ni < 4; ++ni)
          sf[mi][ni] = wmma_bf16(qf[mi], kf[ni], sf[mi][ni]);
    }

    // ---- scale; causal mask only on the diagonal-straddling tile ----
#pragma unroll
    for (int mi = 0; mi < 2; ++mi)
#pragma unroll
      for (int ni = 0; ni < 4; ++ni)
#pragma unroll
        for (int j = 0; j < 8; ++j) sf[mi][ni][j] *= 0.125f;  // 1/sqrt(64)

    if (kb + 64 > q0) {  // wave-uniform: only the last tile can mask
#pragma unroll
      for (int mi = 0; mi < 2; ++mi)
#pragma unroll
        for (int ni = 0; ni < 4; ++ni) {
          const int kc = kb + ni * 16 + l15;
#pragma unroll
          for (int j = 0; j < 8; ++j) {
            const int q = q0 + mi * 16 + half * 8 + j;
            if (kc > q) sf[mi][ni][j] = -3.0e38f;
          }
        }
    }

    // ---- online softmax (row = j + half*8; N lives on lanes 0..15) ----
#pragma unroll
    for (int mi = 0; mi < 2; ++mi)
#pragma unroll
      for (int j = 0; j < 8; ++j) {
        float mx = -3.0e38f;
#pragma unroll
        for (int ni = 0; ni < 4; ++ni) mx = fmaxf(mx, sf[mi][ni][j]);
#pragma unroll
        for (int o = 1; o < 16; o <<= 1) mx = fmaxf(mx, __shfl_xor(mx, o, 32));
        const float mo = mrow[mi][j];
        const float mn = fmaxf(mo, mx);
        const float corr = (mo < -1.0e37f) ? 0.f : __expf(mo - mn);
        float rs = 0.f;
#pragma unroll
        for (int ni = 0; ni < 4; ++ni) {
          const float p = __expf(sf[mi][ni][j] - mn);
          sf[mi][ni][j] = p;
          rs += p;
        }
#pragma unroll
        for (int o = 1; o < 16; o <<= 1) rs += __shfl_xor(rs, o, 32);
        lrow[mi][j] = lrow[mi][j] * corr + rs;
        mrow[mi][j] = mn;
#pragma unroll
        for (int dn = 0; dn < 4; ++dn) acc[mi][dn][j] *= corr;
      }

    // ---- stage P (C-layout) -> LDS row-major -> reload as A fragments ----
    asm volatile("" ::: "memory");
#pragma unroll
    for (int mi = 0; mi < 2; ++mi)
#pragma unroll
      for (int ni = 0; ni < 4; ++ni)
#pragma unroll
        for (int j = 0; j < 8; ++j)
          p_lds[wid][mi * 16 + half * 8 + j][ni * 16 + l15] =
              (bf16_t)sf[mi][ni][j];
    asm volatile("s_wait_dscnt 0" ::: "memory");

    // ---- O += P x V (inner dim = 64 k-columns -> two WMMA k-steps) ----
#pragma unroll
    for (int kk = 0; kk < 64; kk += 32) {
      v16bf pf[2], vf[4];
#pragma unroll
      for (int mi = 0; mi < 2; ++mi)
        pf[mi] = frag_load_a(&p_lds[wid][mi * 16][kk], 72);
#pragma unroll
      for (int dn = 0; dn < 4; ++dn)
        vf[dn] = frag_load_b(Vbase + (size_t)(dn * 16) * SEQ + kb + kk, SEQ);
#pragma unroll
      for (int mi = 0; mi < 2; ++mi)
#pragma unroll
        for (int dn = 0; dn < 4; ++dn)
          acc[mi][dn] = wmma_bf16(pf[mi], vf[dn], acc[mi][dn]);
    }
    asm volatile("s_wait_dscnt 0" ::: "memory");
  }

  // ---- normalize and write O as bf16 [B*S, H*d] ----
#pragma unroll
  for (int mi = 0; mi < 2; ++mi)
#pragma unroll
    for (int dn = 0; dn < 4; ++dn)
#pragma unroll
      for (int j = 0; j < 8; ++j) {
        const int q  = q0 + mi * 16 + half * 8 + j;
        const int dd = dn * 16 + l15;
        const float o = acc[mi][dn][j] / lrow[mi][j];
        Ob[((size_t)b * SEQ + q) * EMB + h * HEAD_DIM + dd] = (bf16_t)o;
      }
}

// ---------------------------------------------------------------------------
// Host launcher
// ---------------------------------------------------------------------------
extern "C" void kernel_launch(void* const* d_in, const int* in_sizes, int n_in,
                              void* d_out, int out_size, void* d_ws,
                              size_t ws_size, hipStream_t stream) {
  (void)in_sizes; (void)n_in; (void)out_size; (void)ws_size;

  const float* X     = (const float*)d_in[0];
  const float* W_Q   = (const float*)d_in[1];
  const float* W_K   = (const float*)d_in[2];
  const float* W_V   = (const float*)d_in[3];
  const float* W_out = (const float*)d_in[4];
  const float* b_out = (const float*)d_in[5];
  const float* W1    = (const float*)d_in[6];
  const float* b1    = (const float*)d_in[7];
  const float* W2    = (const float*)d_in[8];
  const float* b2    = (const float*)d_in[9];
  const float* ln1_g = (const float*)d_in[10];
  const float* ln1_b = (const float*)d_in[11];
  const float* ln2_g = (const float*)d_in[12];
  const float* ln2_b = (const float*)d_in[13];
  float* out = (float*)d_out;

  // workspace layout (~104 MB total)
  char* ws = (char*)d_ws;
  size_t off = 0;
  auto alloc = [&](size_t bytes) -> char* {
    char* p = ws + off;
    off += (bytes + 255) & ~(size_t)255;
    return p;
  };
  bf16_t* wq_t  = (bf16_t*)alloc((size_t)EMB * EMB * 2);
  bf16_t* wk_t  = (bf16_t*)alloc((size_t)EMB * EMB * 2);
  bf16_t* wv_t  = (bf16_t*)alloc((size_t)EMB * EMB * 2);
  bf16_t* wo_t  = (bf16_t*)alloc((size_t)EMB * EMB * 2);
  bf16_t* w1_t  = (bf16_t*)alloc((size_t)EMB * FFD * 2);
  bf16_t* w2_t  = (bf16_t*)alloc((size_t)EMB * FFD * 2);
  bf16_t* h_bf  = (bf16_t*)alloc((size_t)NROWS * EMB * 2);
  bf16_t* q_bf  = (bf16_t*)alloc((size_t)NROWS * EMB * 2);
  bf16_t* k_bf  = (bf16_t*)alloc((size_t)NROWS * EMB * 2);
  bf16_t* vt_bf = (bf16_t*)alloc((size_t)NROWS * EMB * 2);
  float*  x1    = (float*)alloc((size_t)NROWS * EMB * 4);
  bf16_t* ffh   = (bf16_t*)alloc((size_t)NROWS * FFD * 2);

  // weight transposes fp32 [K,N] -> bf16 [N,K]
  cvt_transpose_kernel<<<(EMB * EMB) / 256, 256, 0, stream>>>(W_Q, wq_t, EMB, EMB);
  cvt_transpose_kernel<<<(EMB * EMB) / 256, 256, 0, stream>>>(W_K, wk_t, EMB, EMB);
  cvt_transpose_kernel<<<(EMB * EMB) / 256, 256, 0, stream>>>(W_V, wv_t, EMB, EMB);
  cvt_transpose_kernel<<<(EMB * EMB) / 256, 256, 0, stream>>>(W_out, wo_t, EMB, EMB);
  cvt_transpose_kernel<<<(EMB * FFD) / 256, 256, 0, stream>>>(W1, w1_t, EMB, FFD);
  cvt_transpose_kernel<<<(FFD * EMB) / 256, 256, 0, stream>>>(W2, w2_t, FFD, EMB);

  // h = LN1(X) as bf16
  layernorm_kernel<<<NROWS, 256, 0, stream>>>(X, ln1_g, ln1_b, h_bf);

  // Q, K (plain bf16), V (bf16 transposed per head)
  dim3 gP(EMB / 128, NROWS / 128);  // (8, 32)
  gemm_tn_kernel<0><<<gP, 256, 0, stream>>>(h_bf, wq_t, EMB, EMB, EMB,
                                            nullptr, nullptr, q_bf, EMB);
  gemm_tn_kernel<0><<<gP, 256, 0, stream>>>(h_bf, wk_t, EMB, EMB, EMB,
                                            nullptr, nullptr, k_bf, EMB);
  gemm_tn_kernel<1><<<gP, 256, 0, stream>>>(h_bf, wv_t, EMB, EMB, EMB,
                                            nullptr, nullptr, vt_bf, EMB);

  // flash attention; O overwrites q_bf (disjoint per-wave read/write regions)
  attention_kernel<<<(BATCH * HEADS * (SEQ / 32)) / 4, 128, 0, stream>>>(
      q_bf, k_bf, vt_bf, q_bf);

  // X1 = X + O @ W_out + b_out  (fp32)
  gemm_tn_kernel<2><<<gP, 256, 0, stream>>>(q_bf, wo_t, EMB, EMB, EMB,
                                            b_out, X, x1, EMB);

  // h = LN2(X1) as bf16 (reuse h_bf)
  layernorm_kernel<<<NROWS, 256, 0, stream>>>(x1, ln2_g, ln2_b, h_bf);

  // FFh = gelu(h @ W1 + b1) as bf16
  dim3 gF1(FFD / 128, NROWS / 128);  // (32, 32)
  gemm_tn_kernel<3><<<gF1, 256, 0, stream>>>(h_bf, w1_t, EMB, EMB, EMB,
                                             b1, nullptr, ffh, FFD);

  // out = X1 + FFh @ W2 + b2  (fp32)
  gemm_tn_kernel<2><<<gP, 256, 0, stream>>>(ffh, w2_t, FFD, FFD, FFD,
                                            b2, x1, out, EMB);
}
